// Attention_17918603559191
// MI455X (gfx1250) — matrix operand order repaired and checked by reading, not compiled, not run on hardware
//
#include <hip/hip_runtime.h>
#include <stdint.h>

// ---------------------------------------------------------------------------
// Fixed-point attention for MI455X (gfx1250, wave32, WMMA, TDM double-buffer).
//  B=4, N=2048, D=1024, heads=16, dim_head=64.
// Pipeline: (a) fp32->f16 pre-convert of x / w_qkv / w_proj;
//           (b) QKV GEMM (WMMA f16) -> quantized Q,K,V^T in f16;
//           (c) fused flash attention (online softmax, WMMA f16);
//           (d) output projection GEMM (WMMA f16) + bias -> fp32 out.
// ---------------------------------------------------------------------------

typedef _Float16 f16;
typedef _Float16 v16h __attribute__((ext_vector_type(16)));
typedef _Float16 v8h  __attribute__((ext_vector_type(8)));
typedef float    v8f  __attribute__((ext_vector_type(8)));
typedef float    v4f  __attribute__((ext_vector_type(4)));
typedef unsigned int v4u __attribute__((ext_vector_type(4)));
typedef int      v8i  __attribute__((ext_vector_type(8)));
typedef int      v4i  __attribute__((ext_vector_type(4)));

#if __has_builtin(__builtin_amdgcn_tensor_load_to_lds) && __has_builtin(__builtin_amdgcn_s_wait_tensorcnt)
#define HAVE_TDM 1
#else
#define HAVE_TDM 0
#endif

// ---- reference's make_fxp: clamp to +-(8 - 2^-12), truncate to 2^-12 grid --
__device__ __forceinline__ float make_fxp(float x) {
    const float c = 8.0f - (1.0f / 4096.0f);
    x = fminf(fmaxf(x, -c), c);
    return truncf(x * 4096.0f) * (1.0f / 4096.0f);
}

// ---- WMMA wrapper: D = A(16x32 f16) * B(32x16 f16) + C(16x16 f32) ----------
__device__ __forceinline__ v8f wmma_f16(v16h a, v16h b, v8f c) {
    return __builtin_amdgcn_wmma_f32_16x16x32_f16(false, a, false, b,
                                                  (short)0, c, false, false);
}

// ---- Tensor Data Mover: stage a 2D tile (tile_rows x tile_cols, row-major,
//      row stride in elements) from global into LDS (compacted row-major). ---
__device__ __forceinline__ void tdm_load_2d(uint32_t lds_off, const void* g,
                                            int elem_log2, uint32_t tile_cols,
                                            uint32_t tile_rows,
                                            uint64_t row_stride_elems) {
#if HAVE_TDM
    uint64_t ga = (uint64_t)(uintptr_t)g;
    v4u g0;
    g0[0] = 1u;                                   // count=1, user descriptor
    g0[1] = lds_off;                              // lds_addr (bytes)
    g0[2] = (uint32_t)ga;                         // global_addr[31:0]
    g0[3] = ((uint32_t)(ga >> 32) & 0x01FFFFFFu)  // global_addr[56:32]
            | 0x80000000u;                        // type=2 ("image")
    v8i g1 = {0, 0, 0, 0, 0, 0, 0, 0};
    g1[0] = elem_log2 << 16;                              // data_size
    g1[1] = (int)((row_stride_elems & 0xFFFFu) << 16);    // tensor_dim0 lo16
    g1[2] = (int)((row_stride_elems >> 16) & 0xFFFFu)     // tensor_dim0 hi16
            | 0x40000000;                                 // tensor_dim1 lo16 = 16384
    g1[3] = (int)((tile_cols & 0xFFFFu) << 16);           // tile_dim0
    g1[4] = (int)(tile_rows & 0xFFFFu);                   // tile_dim1 (tile_dim2=0)
    g1[5] = (int)(uint32_t)row_stride_elems;              // dim0_stride lo32
    g1[6] = (int)((row_stride_elems >> 32) & 0xFFFFu);    // dim0_stride hi16
    v4i gz4 = {0, 0, 0, 0};
    v8i gz8 = {0, 0, 0, 0, 0, 0, 0, 0};
    __builtin_amdgcn_tensor_load_to_lds(g0, g1, gz4, gz4, gz8, 0);
#endif
}

// Issue one f16 tile pair (A-tile + B-tile) into LDS. TDM: wave 0 issues,
// no wait here (async). Fallback: cooperative synchronous copy.
__device__ __forceinline__ void issue_pair(f16* ldsA, const f16* gA, int rA,
                                           int cA, long sA, f16* ldsB,
                                           const f16* gB, int rB, int cB,
                                           long sB) {
#if HAVE_TDM
    if (threadIdx.x < 32) {
        tdm_load_2d((uint32_t)(uintptr_t)ldsA, gA, 1, (uint32_t)cA,
                    (uint32_t)rA, (uint64_t)sA);
        tdm_load_2d((uint32_t)(uintptr_t)ldsB, gB, 1, (uint32_t)cB,
                    (uint32_t)rB, (uint64_t)sB);
    }
#else
    for (int i = threadIdx.x; i < rA * cA; i += (int)blockDim.x)
        ldsA[i] = gA[(long)(i / cA) * sA + (i % cA)];
    for (int i = threadIdx.x; i < rB * cB; i += (int)blockDim.x)
        ldsB[i] = gB[(long)(i / cB) * sB + (i % cB)];
#endif
}

// Wait until the *current* tile pair has landed. `more`: another pair is in
// flight behind it (double-buffer steady state). Tensor ops complete in order,
// so TENSORcnt <= 2 means everything older than the newest pair is done.
__device__ __forceinline__ void pair_wait(bool more) {
#if HAVE_TDM
    if (threadIdx.x < 32) {
        if (more) __builtin_amdgcn_s_wait_tensorcnt(2);
        else      __builtin_amdgcn_s_wait_tensorcnt(0);
    }
#endif
    __syncthreads();
}

// ---- fragment builders (gfx1250 WMMA VGPR layouts) -------------------------
// A (16x32 f16): lane -> row m = lane&15; hf = lane>>4;
//   elements 0..15 = K 16*hf..+15 (contiguous)
__device__ __forceinline__ v16h frag_A_f16(const f16* lds16 /*[rows][32]*/,
                                           int m, int hf) {
    v8h c0 = *(const v8h*)(lds16 + m * 32 + 16 * hf);
    v8h c1 = *(const v8h*)(lds16 + m * 32 + 16 * hf + 8);
    v16h a;
#pragma unroll
    for (int i = 0; i < 8; ++i) { a[i] = c0[i]; a[i + 8] = c1[i]; }
    return a;
}
// B (32x16 f16): lane -> col n = lane&15; 16 contiguous K from 16*(lane>>4).
// B tile stored transposed in LDS as [n][K] -> single 32B ds read.
__device__ __forceinline__ v16h frag_B_f16(const f16* lds16 /*[n][32]*/,
                                           int n, int hf) {
    return *(const v16h*)(lds16 + n * 32 + 16 * hf);
}

__device__ __forceinline__ v8f v8f_zero() {
    v8f z = {0.f, 0.f, 0.f, 0.f, 0.f, 0.f, 0.f, 0.f};
    return z;
}

// ===========================================================================
// Kernel 0: fp32 -> f16 pre-convert (one pass; removes all cvt from GEMM hot
// loops and halves staging bandwidth). 8 elements / thread, n % 8 == 0.
// ===========================================================================
__global__ __launch_bounds__(256) void cvt_kernel(const float* __restrict__ s,
                                                  f16* __restrict__ d, int n) {
    int i = (blockIdx.x * 256 + threadIdx.x) * 8;
    if (i + 8 <= n) {
        v4f a = *(const v4f*)(s + i);
        v4f b = *(const v4f*)(s + i + 4);
        v8h o;
#pragma unroll
        for (int j = 0; j < 4; ++j) { o[j] = (f16)a[j]; o[j + 4] = (f16)b[j]; }
        *(v8h*)(d + i) = o;
    }
}

// ===========================================================================
// Kernel 1: qkv = x @ w_qkv^T (f16 WMMA, f32 acc), fxp-quantize, scatter to
//           Q,K ([b,h,n,d]) and V^T ([b,h,d,n]), all f16.
// grid (8192/64, 3072/64), 128 threads (4 waves; wave = 16 rows x 64 cols).
// ===========================================================================
__global__ __launch_bounds__(128) void qkv_kernel(
    const f16* __restrict__ xh, const f16* __restrict__ wh,
    f16* __restrict__ Qh, f16* __restrict__ Kh, f16* __restrict__ Vth) {
    __shared__ __align__(32) f16 ldsA[2][64 * 32];
    __shared__ __align__(32) f16 ldsB[2][64 * 32];
    const int tid = threadIdx.x, lane = tid & 31, wave = tid >> 5;
    const int hf = lane >> 4, lm = lane & 15;
    const int row0 = blockIdx.x * 64;   // token tile
    const int col0 = blockIdx.y * 64;   // output-feature tile
    const f16* gA = xh + (long)row0 * 1024;
    const f16* gB = wh + (long)col0 * 1024;
    v8f acc[4] = {v8f_zero(), v8f_zero(), v8f_zero(), v8f_zero()};

    issue_pair(ldsA[0], gA, 64, 32, 1024, ldsB[0], gB, 64, 32, 1024);
    for (int it = 0; it < 32; ++it) {          // K = 1024, 32 per step
        const int cur = it & 1;
        const bool more = (it + 1) < 32;
        if (more)
            issue_pair(ldsA[cur ^ 1], gA + (it + 1) * 32, 64, 32, 1024,
                       ldsB[cur ^ 1], gB + (it + 1) * 32, 64, 32, 1024);
        pair_wait(more);
        v16h a = frag_A_f16(ldsA[cur], wave * 16 + lm, hf);
#pragma unroll
        for (int nt = 0; nt < 4; ++nt)
            acc[nt] = wmma_f16(a, frag_B_f16(ldsB[cur], nt * 16 + lm, hf), acc[nt]);
        __syncthreads();
    }
#pragma unroll
    for (int nt = 0; nt < 4; ++nt) {
        const int e = col0 + nt * 16 + lm;         // 0..3071
        const int dd = e & 63;
#pragma unroll
        for (int r = 0; r < 8; ++r) {
            const int t = row0 + wave * 16 + 8 * hf + r;  // 0..8191
            const int b = t >> 11, n = t & 2047;
            f16 hv = (f16)make_fxp(acc[nt][r]);
            if (e < 1024) {
                const int h = e >> 6;
                Qh[((long)(b * 16 + h) * 2048 + n) * 64 + dd] = hv;
            } else if (e < 2048) {
                const int h = (e - 1024) >> 6;
                Kh[((long)(b * 16 + h) * 2048 + n) * 64 + dd] = hv;
            } else {
                const int h = (e - 2048) >> 6;
                Vth[((long)(b * 16 + h) * 64 + dd) * 2048 + n] = hv;
            }
        }
    }
}

// ===========================================================================
// Kernel 2: fused flash attention (online softmax). One WG per (b,h,64-query
// block); each wave owns 16 queries (Q fragments pinned in VGPRs). K/V
// streamed in 32-key tiles, TDM double-buffered.
// grid (2048/64, B*heads=64), 128 threads.
// ===========================================================================
__global__ __launch_bounds__(128) void attn_kernel(
    const f16* __restrict__ Qh, const f16* __restrict__ Kh,
    const f16* __restrict__ Vth, f16* __restrict__ Oh) {
    __shared__ __align__(32) f16 ldsK[2][32 * 64];     // [key][d]
    __shared__ __align__(32) f16 ldsV[2][64 * 32];     // [d][key] (V^T tile)
    __shared__ __align__(32) f16 ldsP[4][16 * 32];     // per-wave P transpose
    const int tid = threadIdx.x, lane = tid & 31, wave = tid >> 5;
    const int hf = lane >> 4, lm = lane & 15;
    const int bh = blockIdx.y;                         // b*16 + h
    const f16* Qg = Qh + (long)bh * 2048 * 64;
    const f16* Kg = Kh + (long)bh * 2048 * 64;
    const f16* Vg = Vth + (long)bh * 64 * 2048;
    const int q0 = blockIdx.x * 64 + wave * 16;

    // Q A-fragments stay in registers for the whole kernel (d=64 -> 2 ksteps)
    v16h Aq[2];
#pragma unroll
    for (int s = 0; s < 2; ++s) {
        const f16* p = Qg + (long)(q0 + lm) * 64 + s * 32 + 16 * hf;
        v8h c0 = *(const v8h*)p;
        v8h c1 = *(const v8h*)(p + 8);
#pragma unroll
        for (int i = 0; i < 8; ++i) { Aq[s][i] = c0[i]; Aq[s][i + 8] = c1[i]; }
    }

    v8f Oa[4] = {v8f_zero(), v8f_zero(), v8f_zero(), v8f_zero()};
    float mrow[8], lrow[8];
#pragma unroll
    for (int r = 0; r < 8; ++r) { mrow[r] = -3.0e38f; lrow[r] = 0.f; }
    const float scale = 0.125f;  // d^-0.5

    issue_pair(ldsK[0], Kg, 32, 64, 64, ldsV[0], Vg, 64, 32, 2048);
    for (int it = 0; it < 64; ++it) {            // 2048 keys, 32 per tile
        const int cur = it & 1;
        const bool more = (it + 1) < 64;
        if (more)
            issue_pair(ldsK[cur ^ 1], Kg + (long)(it + 1) * 32 * 64, 32, 64, 64,
                       ldsV[cur ^ 1], Vg + (it + 1) * 32, 64, 32, 2048);
        pair_wait(more);

        // S = Q K^T * scale, fxp-quantized (reference quantizes dots)
        v8f S[2];
#pragma unroll
        for (int nt = 0; nt < 2; ++nt) {
            v8f s = v8f_zero();
#pragma unroll
            for (int ss = 0; ss < 2; ++ss) {
                v16h bfr = *(const v16h*)(ldsK[cur] + (nt * 16 + lm) * 64 +
                                          ss * 32 + 16 * hf);
                s = wmma_f16(Aq[ss], bfr, s);
            }
#pragma unroll
            for (int i = 0; i < 8; ++i) s[i] = make_fxp(s[i] * scale);
            S[nt] = s;
        }
        // online softmax: row reductions across the 16-lane half-wave
        float rmax[8];
#pragma unroll
        for (int r = 0; r < 8; ++r) rmax[r] = fmaxf(S[0][r], S[1][r]);
#pragma unroll
        for (int msk = 1; msk < 16; msk <<= 1)
#pragma unroll
            for (int r = 0; r < 8; ++r)
                rmax[r] = fmaxf(rmax[r], __shfl_xor(rmax[r], msk, 32));
        float alpha[8];
#pragma unroll
        for (int r = 0; r < 8; ++r) {
            float mn = fmaxf(mrow[r], rmax[r]);
            alpha[r] = __expf(mrow[r] - mn);
            mrow[r] = mn;
        }
        float rsum[8];
#pragma unroll
        for (int r = 0; r < 8; ++r) rsum[r] = 0.f;
#pragma unroll
        for (int nt = 0; nt < 2; ++nt)
#pragma unroll
            for (int r = 0; r < 8; ++r) {
                float p = __expf(S[nt][r] - mrow[r]);
                S[nt][r] = p;
                rsum[r] += p;
            }
#pragma unroll
        for (int msk = 1; msk < 16; msk <<= 1)
#pragma unroll
            for (int r = 0; r < 8; ++r) rsum[r] += __shfl_xor(rsum[r], msk, 32);
#pragma unroll
        for (int r = 0; r < 8; ++r) lrow[r] = lrow[r] * alpha[r] + rsum[r];
#pragma unroll
        for (int dt = 0; dt < 4; ++dt)
#pragma unroll
            for (int r = 0; r < 8; ++r) Oa[dt][r] *= alpha[r];

        // transpose P (C-layout -> A-layout) through per-wave LDS slab
        f16* Pw = ldsP[wave];
#pragma unroll
        for (int nt = 0; nt < 2; ++nt)
#pragma unroll
            for (int r = 0; r < 8; ++r)
                Pw[(8 * hf + r) * 32 + nt * 16 + lm] = (f16)S[nt][r];
        asm volatile("s_wait_dscnt 0x0" ::: "memory");  // same-wave LDS RAW
        v16h Ap;
        {
            const f16* p = Pw + lm * 32 + 16 * hf;
            v8h c0 = *(const v8h*)p;
            v8h c1 = *(const v8h*)(p + 8);
#pragma unroll
            for (int i = 0; i < 8; ++i) { Ap[i] = c0[i]; Ap[i + 8] = c1[i]; }
        }
        // O += P @ V   (B-fragments = contiguous rows of V^T tile)
#pragma unroll
        for (int dt = 0; dt < 4; ++dt) {
            v16h bfr = *(const v16h*)(ldsV[cur] + (dt * 16 + lm) * 32 + 16 * hf);
            Oa[dt] = wmma_f16(Ap, bfr, Oa[dt]);
        }
        __syncthreads();
    }

    // normalize, quantize, store O in [b, n, h*64+dd] layout for the proj GEMM
    const int b = bh >> 4, h = bh & 15;
#pragma unroll
    for (int dt = 0; dt < 4; ++dt)
#pragma unroll
        for (int r = 0; r < 8; ++r) {
            const int n = q0 + 8 * hf + r;
            float v = make_fxp(Oa[dt][r] / lrow[r]);
            Oh[(long)(b * 2048 + n) * 1024 + h * 64 + dt * 16 + lm] = (f16)v;
        }
}

// ===========================================================================
// Kernel 3: out = fxp(O @ w_proj^T + b_proj), fp32 result.
// grid (8192/64, 1024/64), 128 threads.
// ===========================================================================
__global__ __launch_bounds__(128) void proj_kernel(
    const f16* __restrict__ Oh, const f16* __restrict__ wh,
    const float* __restrict__ bias, float* __restrict__ out) {
    __shared__ __align__(32) f16 ldsA[2][64 * 32];
    __shared__ __align__(32) f16 ldsB[2][64 * 32];
    const int tid = threadIdx.x, lane = tid & 31, wave = tid >> 5;
    const int hf = lane >> 4, lm = lane & 15;
    const int row0 = blockIdx.x * 64;
    const int col0 = blockIdx.y * 64;
    const f16* gA = Oh + (long)row0 * 1024;
    const f16* gB = wh + (long)col0 * 1024;
    v8f acc[4] = {v8f_zero(), v8f_zero(), v8f_zero(), v8f_zero()};

    issue_pair(ldsA[0], gA, 64, 32, 1024, ldsB[0], gB, 64, 32, 1024);
    for (int it = 0; it < 32; ++it) {
        const int cur = it & 1;
        const bool more = (it + 1) < 32;
        if (more)
            issue_pair(ldsA[cur ^ 1], gA + (it + 1) * 32, 64, 32, 1024,
                       ldsB[cur ^ 1], gB + (it + 1) * 32, 64, 32, 1024);
        pair_wait(more);
        v16h a = frag_A_f16(ldsA[cur], wave * 16 + lm, hf);
#pragma unroll
        for (int nt = 0; nt < 4; ++nt)
            acc[nt] = wmma_f16(a, frag_B_f16(ldsB[cur], nt * 16 + lm, hf), acc[nt]);
        __syncthreads();
    }
#pragma unroll
    for (int nt = 0; nt < 4; ++nt) {
        const int e = col0 + nt * 16 + lm;
        const float bb = bias[e];
#pragma unroll
        for (int r = 0; r < 8; ++r) {
            const int t = row0 + wave * 16 + 8 * hf + r;
            out[(long)t * 1024 + e] = make_fxp(acc[nt][r] + bb);
        }
    }
}

// ===========================================================================
extern "C" void kernel_launch(void* const* d_in, const int* in_sizes, int n_in,
                              void* d_out, int out_size, void* d_ws,
                              size_t ws_size, hipStream_t stream) {
    const float* x      = (const float*)d_in[0];
    const float* w_qkv  = (const float*)d_in[1];
    const float* w_proj = (const float*)d_in[2];
    const float* b_proj = (const float*)d_in[3];
    float* out = (float*)d_out;

    const size_t HD = (size_t)4 * 16 * 2048 * 64;  // 8,388,608 elems per tensor
    f16* Qh   = (f16*)d_ws;
    f16* Kh   = Qh + HD;
    f16* Vth  = Kh + HD;
    f16* Oh   = Vth + HD;            // [B, N, 1024] f16
    f16* xh   = Oh + HD;             // x in f16        (8,388,608)
    f16* wqh  = xh + HD;             // w_qkv in f16    (3,145,728)
    f16* wph  = wqh + (size_t)3072 * 1024;  // w_proj f16 (1,048,576)
    // total workspace use: ~92 MB

    cvt_kernel<<<dim3((int)(HD / 2048)), 256, 0, stream>>>(x, xh, (int)HD);
    cvt_kernel<<<dim3(3072 * 1024 / 2048), 256, 0, stream>>>(w_qkv, wqh, 3072 * 1024);
    cvt_kernel<<<dim3(1024 * 1024 / 2048), 256, 0, stream>>>(w_proj, wph, 1024 * 1024);

    qkv_kernel<<<dim3(8192 / 64, 3072 / 64), 128, 0, stream>>>(xh, wqh, Qh, Kh, Vth);
    attn_kernel<<<dim3(2048 / 64, 64), 128, 0, stream>>>(Qh, Kh, Vth, Oh);
    proj_kernel<<<dim3(8192 / 64, 1024 / 64), 128, 0, stream>>>(Oh, wph, b_proj, out);
    (void)in_sizes; (void)n_in; (void)out_size; (void)ws_size;
}
